// MultiHeadAttention_66872640799208
// MI455X (gfx1250) — compile-verified, matrix operand order checked by reading, not run on hardware
//
#include <hip/hip_runtime.h>
#include <hip/hip_bf16.h>
#include <math.h>

// ---------------------------------------------------------------------------
// MI455X (gfx1250) multi-head-attention-ish pipeline.
// All GEMMs use v_wmma_f32_16x16x32_bf16 (wave32, f32 accum).
// The bf16 GEMMs (scores, attn@V) pull tiles with the Tensor Data Mover,
// double-buffered on TENSORcnt in scores only.
// ---------------------------------------------------------------------------

typedef __bf16 v16bf __attribute__((ext_vector_type(16)));
typedef float  v8f   __attribute__((ext_vector_type(8)));
typedef unsigned int u32x4 __attribute__((ext_vector_type(4)));
typedef int          i32x8 __attribute__((ext_vector_type(8)));
typedef int          i32x4 __attribute__((ext_vector_type(4)));

static constexpr int Bb = 8;      // batch
static constexpr int Ss = 2048;   // sequence
static constexpr int Dd = 1024;   // model dim

#define WMMA_BF16(a, b, c) \
    __builtin_amdgcn_wmma_f32_16x16x32_bf16(false, (a), false, (b), (short)0, (c), false, false)

// Issue one TDM 2-D tile load (bf16 elements): tile_x elems wide, tile_y rows,
// row pitch stride_elems. Descriptor packed per CDNA5 ISA.
// EXEC is ignored by TENSOR_LOAD_TO_LDS: call it from exactly one wave.
__device__ __forceinline__ void tdm_load_tile_bf16(unsigned lds_off,
                                                   const __bf16* gptr,
                                                   unsigned tile_x, unsigned tile_y,
                                                   unsigned long long stride_elems) {
  unsigned long long ga = (unsigned long long)(size_t)gptr;
  u32x4 g0;
  g0[0] = 1u;                                            // count=1, user descriptor
  g0[1] = lds_off;                                       // lds_addr (bytes)
  g0[2] = (unsigned)(ga & 0xFFFFFFFFull);                // global_addr[31:0]
  g0[3] = (unsigned)((ga >> 32) & 0x01FFFFFFull) | (2u << 30);  // addr[56:32] | type=2
  i32x8 g1;
  g1[0] = (int)(1u << 16);                               // wg_mask=0, data_size=2B
  g1[1] = (int)((tile_x & 0xFFFFu) << 16);               // tensor_dim0[15:0]
  g1[2] = (int)((tile_x >> 16) | ((tile_y & 0xFFFFu) << 16));   // dim0 hi | dim1 lo
  g1[3] = (int)(((tile_y >> 16) & 0xFFFFu) | ((tile_x & 0xFFFFu) << 16)); // dim1 hi | tile_dim0
  g1[4] = (int)(tile_y & 0xFFFFu);                       // tile_dim1, tile_dim2=0
  g1[5] = (int)(unsigned)(stride_elems & 0xFFFFFFFFull); // dim0_stride[31:0]
  g1[6] = (int)(unsigned)((stride_elems >> 32) & 0xFFFFull); // dim0_stride[47:32]
  g1[7] = 0;
  i32x4 z4 = {0, 0, 0, 0};
#if __clang_major__ >= 23
  i32x8 z8 = {0, 0, 0, 0, 0, 0, 0, 0};
  __builtin_amdgcn_tensor_load_to_lds(g0, g1, z4, z4, z8, 0);
#else
  __builtin_amdgcn_tensor_load_to_lds(g0, g1, z4, z4, 0);
#endif
}

// ---------------------------------------------------------------------------
// Kernel 1: Y[m,n] = sum_d X[m,d] * W[n,d] + bias[n]   (f32 in, bf16 out)
// Block: 256 thr (8 waves), tile M=64 x N=128, K-step 32.
// Wave w: M-subtile (w&3), N-half (w>>2), 4 accumulators of 16x16.
// ---------------------------------------------------------------------------
__global__ __launch_bounds__(256) void proj_gemm_kernel(
    const float* __restrict__ X, const float* __restrict__ W,
    const float* __restrict__ bias, __bf16* __restrict__ Y,
    int M, int N, int K) {
  __shared__ __bf16 ldsA[64][32];
  __shared__ __bf16 ldsB[128][32];

  const int m0 = blockIdx.x * 64;
  const int n0 = blockIdx.y * 128;
  const int t = threadIdx.x;
  const int w = t >> 5, lane = t & 31;
  const int lr = lane & 15, kh = (lane >> 4) * 16;
  const int mt = w & 3, ng = w >> 2;

  v8f acc[4] = {};

  for (int k0 = 0; k0 < K; k0 += 32) {
    // stage A: 64x32 f32 -> bf16
    {
      const int idx = t * 8;
      const int r = idx >> 5, c = idx & 31;
      const float* src = &X[(size_t)(m0 + r) * K + k0 + c];
      if (k0 + 32 < K) __builtin_prefetch(src + 32, 0, 1);  // global_prefetch_b8
      float4 f0 = *(const float4*)(src);
      float4 f1 = *(const float4*)(src + 4);
      __bf16* d = &ldsA[r][c];
      d[0] = (__bf16)f0.x; d[1] = (__bf16)f0.y; d[2] = (__bf16)f0.z; d[3] = (__bf16)f0.w;
      d[4] = (__bf16)f1.x; d[5] = (__bf16)f1.y; d[6] = (__bf16)f1.z; d[7] = (__bf16)f1.w;
    }
    // stage B: 128x32 f32 -> bf16 (W rows are already the "transposed" operand)
    {
      const int idx = t * 16;
      const int r = idx >> 5, c = idx & 31;
      const float* src = &W[(size_t)(n0 + r) * K + k0 + c];
      if (k0 + 32 < K) __builtin_prefetch(src + 32, 0, 1);
      __bf16* d = &ldsB[r][c];
#pragma unroll
      for (int j = 0; j < 4; ++j) {
        float4 f = *(const float4*)(src + 4 * j);
        d[4 * j + 0] = (__bf16)f.x; d[4 * j + 1] = (__bf16)f.y;
        d[4 * j + 2] = (__bf16)f.z; d[4 * j + 3] = (__bf16)f.w;
      }
    }
    __syncthreads();
    v16bf a = *(const v16bf*)&ldsA[mt * 16 + lr][kh];
#pragma unroll
    for (int j = 0; j < 4; ++j) {
      v16bf b = *(const v16bf*)&ldsB[ng * 64 + j * 16 + lr][kh];
      acc[j] = WMMA_BF16(a, b, acc[j]);
    }
    __syncthreads();
  }

#pragma unroll
  for (int j = 0; j < 4; ++j) {
#pragma unroll
    for (int r = 0; r < 8; ++r) {
      const int mm = m0 + mt * 16 + r + ((lane >> 4) << 3);
      const int nn = n0 + ng * 64 + j * 16 + lr;
      Y[(size_t)mm * N + nn] = (__bf16)(acc[j][r] + bias[nn]);
    }
  }
}

// ---------------------------------------------------------------------------
// Kernel 2: scores[b,q,k] = (1/D) * sum_d qp[b,q,d] * kp[b,k,d]
// bf16 x bf16 -> f32. Tiles come in via the Tensor Data Mover, ping-pong
// buffered: issue next pair, then s_wait_tensorcnt(2) -- TDM ops of one wave
// complete in order, so <=2 outstanding means the current pair has landed.
// Block tile 64(q) x 64(k), K-step 32.
// ---------------------------------------------------------------------------
__global__ __launch_bounds__(256) void scores_kernel(
    const __bf16* __restrict__ qp, const __bf16* __restrict__ kp,
    float* __restrict__ scores, float inv_dk) {
  __shared__ __bf16 ldsA[2][64][32];
  __shared__ __bf16 ldsB[2][64][32];

  const int b = blockIdx.z;
  const int q0 = blockIdx.x * 64;
  const int k0n = blockIdx.y * 64;
  const int t = threadIdx.x;
  const int w = t >> 5, lane = t & 31;
  const int lr = lane & 15, kh = (lane >> 4) * 16;
  const int mt = w & 3, nh = w >> 2;

  const __bf16* qbase = qp + ((size_t)b * Ss + q0) * Dd;
  const __bf16* kbase = kp + ((size_t)b * Ss + k0n) * Dd;
  const unsigned offA0 = (unsigned)(size_t)&ldsA[0][0][0];
  const unsigned offA1 = (unsigned)(size_t)&ldsA[1][0][0];
  const unsigned offB0 = (unsigned)(size_t)&ldsB[0][0][0];
  const unsigned offB1 = (unsigned)(size_t)&ldsB[1][0][0];

  v8f acc[2] = {};

  if (t < 32) {  // one wave drives the TDM; TENSORcnt is per-wave
    tdm_load_tile_bf16(offA0, qbase, 32u, 64u, (unsigned long long)Dd);
    tdm_load_tile_bf16(offB0, kbase, 32u, 64u, (unsigned long long)Dd);
  }

  const int NIT = Dd / 32;
  for (int it = 0; it < NIT; ++it) {
    const int cur = it & 1;
    if (t < 32) {
      if (it + 1 < NIT) {  // prefetch next tile pair into the other buffer
        const int k1 = (it + 1) * 32;
        tdm_load_tile_bf16(cur ? offA0 : offA1, qbase + k1, 32u, 64u,
                           (unsigned long long)Dd);
        tdm_load_tile_bf16(cur ? offB0 : offB1, kbase + k1, 32u, 64u,
                           (unsigned long long)Dd);
        __builtin_amdgcn_s_wait_tensorcnt(2);  // current pair done, next in flight
      } else {
        __builtin_amdgcn_s_wait_tensorcnt(0);
      }
    }
    __syncthreads();
    v16bf a = *(const v16bf*)&ldsA[cur][mt * 16 + lr][kh];
#pragma unroll
    for (int j = 0; j < 2; ++j) {
      v16bf bf = *(const v16bf*)&ldsB[cur][nh * 32 + j * 16 + lr][kh];
      acc[j] = WMMA_BF16(a, bf, acc[j]);
    }
    __syncthreads();
  }

#pragma unroll
  for (int j = 0; j < 2; ++j) {
#pragma unroll
    for (int r = 0; r < 8; ++r) {
      const int qq = q0 + mt * 16 + r + ((lane >> 4) << 3);
      const int kk = k0n + nh * 32 + j * 16 + lr;
      scores[((size_t)b * Ss + qq) * Ss + kk] = acc[j][r] * inv_dk;
    }
  }
}

// ---------------------------------------------------------------------------
// Kernel 3: colsum[b,k] = sum_q exp(scores[b,q,k])   (softmax over axis=1)
// Values are O(1/32) so no max-subtraction is needed.
// ---------------------------------------------------------------------------
__global__ __launch_bounds__(256) void colsum_kernel(
    const float* __restrict__ scores, float* __restrict__ colsum) {
  const int b = blockIdx.x >> 3;
  const int k = ((blockIdx.x & 7) << 8) + threadIdx.x;
  const float* base = scores + (size_t)b * Ss * Ss + k;
  float s = 0.f;
  for (int q = 0; q < Ss; ++q) s += __expf(base[(size_t)q * Ss]);
  colsum[b * Ss + k] = s;
}

// Kernel 4: P[b,q,k] = exp(scores)/colsum  -> bf16
__global__ __launch_bounds__(256) void pnorm_kernel(
    const float* __restrict__ scores, const float* __restrict__ colsum,
    __bf16* __restrict__ P, long long n) {
  long long i = (long long)blockIdx.x * 256 + threadIdx.x;
  const long long stride = (long long)gridDim.x * 256;
  for (; i < n; i += stride) {
    const int k = (int)(i & (Ss - 1));
    const int b = (int)(i >> 22);  // S*S = 2^22
    P[i] = (__bf16)(__expf(scores[i]) / colsum[b * Ss + k]);
  }
}

// ---------------------------------------------------------------------------
// Kernel 5: attn_raw[b,q,d] = sum_k P[b,q,k] * vp[b,k,d]
// A tile (P, row-major) comes in via TDM; the vp B operand needs
// column-major in LDS -> cooperative transpose during staging.
// ---------------------------------------------------------------------------
__global__ __launch_bounds__(256) void attn_v_kernel(
    const __bf16* __restrict__ P, const __bf16* __restrict__ vp,
    float* __restrict__ attn_raw) {
  __shared__ __bf16 ldsA[64][32];   // P tile  [q][k]
  __shared__ __bf16 ldsBt[64][32];  // vp tile [d][k] (transposed)

  const int b = blockIdx.z;
  const int q0 = blockIdx.x * 64;
  const int d0 = blockIdx.y * 64;
  const int t = threadIdx.x;
  const int w = t >> 5, lane = t & 31;
  const int lr = lane & 15, kh = (lane >> 4) * 16;
  const int mt = w & 3, nh = w >> 2;

  const __bf16* pbase = P + ((size_t)b * Ss + q0) * Ss;
  const unsigned offA = (unsigned)(size_t)&ldsA[0][0];

  v8f acc[2] = {};

  for (int k0 = 0; k0 < Ss; k0 += 32) {
    if (t < 32) {  // A tile via TDM
      tdm_load_tile_bf16(offA, pbase + k0, 32u, 64u, (unsigned long long)Ss);
      __builtin_amdgcn_s_wait_tensorcnt(0);
    }
    {  // B^T: vp[b][k0..+31][d0..+63] -> ldsBt[d][k]
      const int idx = t * 8;
      const int kk = idx >> 6, nn = idx & 63;
      const __bf16* src = &vp[((size_t)b * Ss + k0 + kk) * Dd + d0 + nn];
#pragma unroll
      for (int j2 = 0; j2 < 8; ++j2) ldsBt[nn + j2][kk] = src[j2];
    }
    __syncthreads();
    v16bf a = *(const v16bf*)&ldsA[mt * 16 + lr][kh];
#pragma unroll
    for (int j = 0; j < 2; ++j) {
      v16bf bf = *(const v16bf*)&ldsBt[nh * 32 + j * 16 + lr][kh];
      acc[j] = WMMA_BF16(a, bf, acc[j]);
    }
    __syncthreads();
  }

#pragma unroll
  for (int j = 0; j < 2; ++j) {
#pragma unroll
    for (int r = 0; r < 8; ++r) {
      const int qq = q0 + mt * 16 + r + ((lane >> 4) << 3);
      const int dd = d0 + nh * 32 + j * 16 + lr;
      attn_raw[((size_t)b * Ss + qq) * Dd + dd] = acc[j][r];
    }
  }
}

// ---------------------------------------------------------------------------
// Kernel 6: attn_w = softmax over s;  attn_out = attn_raw + residual(q)
// ---------------------------------------------------------------------------
__global__ __launch_bounds__(256) void out_kernel(
    const float* __restrict__ attn_raw, const float* __restrict__ q_in,
    float* __restrict__ out_attn, float* __restrict__ out_w) {
  const int b = blockIdx.x >> 2;
  const int d = ((blockIdx.x & 3) << 8) + threadIdx.x;
  const float* base = attn_raw + (size_t)b * Ss * Dd + d;
  float s = 0.f;
  for (int i = 0; i < Ss; ++i) s += __expf(base[(size_t)i * Dd]);
  const float inv = 1.f / s;
  const float* rbase = q_in + (size_t)b * Ss * Dd + d;
  float* oa = out_attn + (size_t)b * Ss * Dd + d;
  float* ow = out_w + (size_t)b * Ss * Dd + d;
  for (int i = 0; i < Ss; ++i) {
    const float x = base[(size_t)i * Dd];
    oa[(size_t)i * Dd] = x + rbase[(size_t)i * Dd];
    ow[(size_t)i * Dd] = __expf(x) * inv;
  }
}

// ---------------------------------------------------------------------------
extern "C" void kernel_launch(void* const* d_in, const int* in_sizes, int n_in,
                              void* d_out, int out_size, void* d_ws, size_t ws_size,
                              hipStream_t stream) {
  (void)in_sizes; (void)n_in; (void)out_size; (void)ws_size;
  const float* q  = (const float*)d_in[0];
  const float* k  = (const float*)d_in[1];
  const float* v  = (const float*)d_in[2];
  const float* Wq = (const float*)d_in[3];
  const float* bq = (const float*)d_in[4];
  const float* Wk = (const float*)d_in[5];
  const float* bk = (const float*)d_in[6];
  const float* Wv = (const float*)d_in[7];
  const float* bv = (const float*)d_in[8];

  // Workspace layout (bytes). Regions reused once dead:
  //   [  0, 32M) qp  bf16     -> later reused for P (with kp region)
  //   [ 32M,64M) kp  bf16
  //   [ 64M,96M) vp  bf16
  //   [ 96M,224M) scores f32  -> later reused for attn_raw f32
  //   [224M,224M+64K) colsum f32
  char* ws = (char*)d_ws;
  const size_t MB = 1ull << 20;
  __bf16* qp = (__bf16*)(ws + 0 * MB);
  __bf16* kp = (__bf16*)(ws + 32 * MB);
  __bf16* vp = (__bf16*)(ws + 64 * MB);
  float* scores = (float*)(ws + 96 * MB);
  float* colsum = (float*)(ws + 224 * MB);
  __bf16* P = (__bf16*)(ws + 0 * MB);        // reuse qp+kp (64MB)
  float* attn_raw = (float*)(ws + 96 * MB);  // reuse scores (64MB of 128MB)

  float* out_attn = (float*)d_out;
  float* out_w = (float*)d_out + (size_t)Bb * Ss * Dd;

  const dim3 blk(256);
  const int Mtot = Bb * Ss;  // 16384

  // 1) projections
  {
    dim3 g(Mtot / 64, Dd / 128);
    proj_gemm_kernel<<<g, blk, 0, stream>>>(q, Wq, bq, qp, Mtot, Dd, Dd);
    proj_gemm_kernel<<<g, blk, 0, stream>>>(k, Wk, bk, kp, Mtot, Dd, Dd);
    proj_gemm_kernel<<<g, blk, 0, stream>>>(v, Wv, bv, vp, Mtot, Dd, Dd);
  }
  // 2) scores = qp @ kp^T / D
  {
    dim3 g(Ss / 64, Ss / 64, Bb);
    scores_kernel<<<g, blk, 0, stream>>>(qp, kp, scores, 1.0f / (float)Dd);
  }
  // 3) column softmax over q
  colsum_kernel<<<dim3(Bb * (Ss / 256)), blk, 0, stream>>>(scores, colsum);
  pnorm_kernel<<<dim3(4096), blk, 0, stream>>>(scores, colsum, P,
                                               (long long)Bb * Ss * Ss);
  // 4) attn_raw = P @ vp
  {
    dim3 g(Ss / 64, Dd / 64, Bb);
    attn_v_kernel<<<g, blk, 0, stream>>>(P, vp, attn_raw);
  }
  // 5) sequence softmax + residual
  out_kernel<<<dim3(Bb * (Dd / 256)), blk, 0, stream>>>(attn_raw, q, out_attn, out_w);
}
